// AnaXnetGCN_5970004541694
// MI455X (gfx1250) — matrix operand order repaired and checked by reading, not compiled, not run on hardware
//
#include <hip/hip_runtime.h>
#include <hip/hip_bf16.h>

// ---------------------------------------------------------------------------
// Problem dims (compile-time)
// ---------------------------------------------------------------------------
#define Bsz   2048
#define Nn    18
#define D_IN  1024
#define D_HID 2048
#define D_OUT 1024
#define NCLS  14
#define Mrows (Bsz * Nn)   // 36864

typedef __attribute__((ext_vector_type(8)))  float  v8f;
typedef __attribute__((ext_vector_type(16))) __bf16 v16bf;

union Frag16 { uint4 q[2]; v16bf v; };

__device__ __forceinline__ unsigned short f2bf(float f) {
  unsigned int x = __float_as_uint(f);
  x += 0x7FFFu + ((x >> 16) & 1u);          // round-to-nearest-even
  return (unsigned short)(x >> 16);
}
__device__ __forceinline__ unsigned int pack2bf(float a, float b) {
  return (unsigned int)f2bf(a) | ((unsigned int)f2bf(b) << 16);
}
__device__ __forceinline__ float bf2f(unsigned short h) {
  return __uint_as_float(((unsigned int)h) << 16);
}
__device__ __forceinline__ float waveReduce(float v) {
  #pragma unroll
  for (int o = 16; o > 0; o >>= 1) v += __shfl_xor(v, o, 32);
  return v;
}

// CDNA5 async global->LDS DMA (no data VGPRs, tracked by ASYNCcnt)
__device__ __forceinline__ void async_b128(unsigned lds_off, const void* gptr) {
  asm volatile("global_load_async_to_lds_b128 %0, %1, off"
               :: "v"(lds_off), "v"(gptr) : "memory");
}
__device__ __forceinline__ void wait_async0() {
  asm volatile("s_wait_asynccnt 0x0" ::: "memory");
}

// ---------------------------------------------------------------------------
// Elementwise fp32 -> bf16 (8 elems / thread, vectorized)
// ---------------------------------------------------------------------------
__global__ __launch_bounds__(256)
void convert_bf16_kernel(const float* __restrict__ src,
                         unsigned short* __restrict__ dst, long long n8)
{
  const long long i = (long long)blockIdx.x * 256 + threadIdx.x;
  if (i >= n8) return;
  const float4* s = (const float4*)(src + i * 8);
  const float4 a = s[0], b = s[1];
  uint4 o;
  o.x = pack2bf(a.x, a.y); o.y = pack2bf(a.z, a.w);
  o.z = pack2bf(b.x, b.y); o.w = pack2bf(b.z, b.w);
  ((uint4*)dst)[i] = o;
}

// ---------------------------------------------------------------------------
// W[K][N] fp32 -> Wt[N][K] bf16 (32x32 tiles through LDS)
// ---------------------------------------------------------------------------
__global__ __launch_bounds__(256)
void transpose_convert_kernel(const float* __restrict__ W,
                              unsigned short* __restrict__ Wt, int K, int N)
{
  __shared__ float tile[32][33];
  const int tx = threadIdx.x & 31, ty = threadIdx.x >> 5;  // 32 x 8
  const int n0 = blockIdx.x * 32, k0 = blockIdx.y * 32;
  #pragma unroll
  for (int i = 0; i < 32; i += 8)
    tile[ty + i][tx] = W[(size_t)(k0 + ty + i) * N + n0 + tx];
  __syncthreads();
  #pragma unroll
  for (int i = 0; i < 32; i += 8)
    Wt[(size_t)(n0 + ty + i) * K + k0 + tx] = f2bf(tile[tx][ty + i]);
}

// ---------------------------------------------------------------------------
// Tiled bf16 WMMA GEMM:  C[M,N](bf16) = A[M,K](bf16) x Bt[N,K](bf16)
// 256 thr (8 waves), tile 128x128, BK=64 (16 WMMAs / stage).
// Double-buffered LDS fed by global_load_async_to_lds_b128 (ASYNCcnt):
// no staging VGPRs -> no spills; DMA of tile k+1 overlaps WMMAs of tile k.
// ---------------------------------------------------------------------------
__global__ __launch_bounds__(256)
void gemm_wmma_bf16(const unsigned short* __restrict__ A,
                    const unsigned short* __restrict__ Bt,
                    unsigned short* __restrict__ Cout,
                    int M, int N, int K)
{
  constexpr int BM = 128, BN = 128, BK = 64, LD = BK + 8; // pad: conflict-free frags
  __shared__ alignas(16) unsigned short As[2][BM * LD];
  __shared__ alignas(16) unsigned short Bs[2][BN * LD];

  const int t    = threadIdx.x;
  const int lane = t & 31;
  const int wave = t >> 5;
  const int wm   = wave & 3;     // 4 wave-rows of 32
  const int wn   = wave >> 2;    // 2 wave-cols of 64
  const int lm   = lane & 15;
  const int lhi  = lane >> 4;    // 0/1 half of wave

  const int row0 = blockIdx.y * BM;
  const int col0 = blockIdx.x * BN;

  // Loader: 2 threads per row, 32 halves (64B = 4 x b128 DMA) each.
  const int l_row  = t >> 1;
  const int l_half = t & 1;

  v8f acc[2][4];
  #pragma unroll
  for (int i = 0; i < 2; ++i)
    #pragma unroll
    for (int j = 0; j < 4; ++j) acc[i][j] = (v8f)0.0f;

  const int nk = K / BK;

  auto stageAsync = [&](int kt, int buf) {
    const unsigned short* ga = A  + (size_t)(row0 + l_row) * K + kt * BK + l_half * 32;
    const unsigned short* gb = Bt + (size_t)(col0 + l_row) * K + kt * BK + l_half * 32;
    const unsigned la = (unsigned)(size_t)&As[buf][l_row * LD + l_half * 32];
    const unsigned lb = (unsigned)(size_t)&Bs[buf][l_row * LD + l_half * 32];
    #pragma unroll
    for (int i = 0; i < 4; ++i) {
      async_b128(la + i * 16, ga + i * 8);
      async_b128(lb + i * 16, gb + i * 8);
    }
  };

  auto compute = [&](int buf) {
    #pragma unroll
    for (int kk = 0; kk < 2; ++kk) {
      // A frag (16-bit 16x32 layout): lane<16 -> K{0..7,16..23}; lane>=16 -> K{8..15,24..31}
      Frag16 af[2], bfr[4];
      #pragma unroll
      for (int mi = 0; mi < 2; ++mi) {
        const unsigned short* p = &As[buf][(wm * 32 + mi * 16 + lm) * LD + kk * 32 + lhi * 8];
        af[mi].q[0] = *(const uint4*)p;
        af[mi].q[1] = *(const uint4*)(p + 16);
      }
      // B frag (32x16): lane<16 -> K{0..7,16..23} of column n; lane>=16 -> K{8..15,24..31}
      #pragma unroll
      for (int ni = 0; ni < 4; ++ni) {
        const unsigned short* p = &Bs[buf][(wn * 64 + ni * 16 + lm) * LD + kk * 32 + lhi * 8];
        bfr[ni].q[0] = *(const uint4*)p;
        bfr[ni].q[1] = *(const uint4*)(p + 16);
      }
      #pragma unroll
      for (int mi = 0; mi < 2; ++mi)
        #pragma unroll
        for (int ni = 0; ni < 4; ++ni)
          acc[mi][ni] = __builtin_amdgcn_wmma_f32_16x16x32_bf16(
              false, af[mi].v, false, bfr[ni].v, (short)0, acc[mi][ni], false, false);
    }
  };

  // prologue: DMA tile 0 into buf0
  stageAsync(0, 0);
  wait_async0();
  __syncthreads();

  for (int kt = 0; kt < nk; kt += 2) {
    // compute buf0, DMA kt+1 -> buf1
    if (kt + 1 < nk) stageAsync(kt + 1, 1);
    compute(0);
    wait_async0();
    __syncthreads();
    // compute buf1, DMA kt+2 -> buf0
    if (kt + 1 < nk) {
      if (kt + 2 < nk) stageAsync(kt + 2, 0);
      compute(1);
      wait_async0();
      __syncthreads();
    }
  }

  // C layout: lane<16: VGPR r -> (M=r, N=lane); lane>=16: (M=8+r, N=lane-16)
  #pragma unroll
  for (int mi = 0; mi < 2; ++mi)
    #pragma unroll
    for (int ni = 0; ni < 4; ++ni) {
      const int gm0 = row0 + wm * 32 + mi * 16 + lhi * 8;
      const int gn  = col0 + wn * 64 + ni * 16 + lm;
      #pragma unroll
      for (int r = 0; r < 8; ++r)
        Cout[(size_t)(gm0 + r) * N + gn] = f2bf(acc[mi][ni][r]);
    }
}

// ---------------------------------------------------------------------------
// x[b,i,k] = LeakyReLU_0.2( sum_j adj[i,j] * h1[b,j,k] )   (bf16 in/out)
// grid = (D/256, B); every h1 element read exactly once.
// ---------------------------------------------------------------------------
__global__ __launch_bounds__(256)
void adj_lrelu_kernel(const unsigned short* __restrict__ h1,
                      const float* __restrict__ adj,
                      unsigned short* __restrict__ xout, int D)
{
  __shared__ float adjs[Nn * Nn];
  const int t = threadIdx.x;
  for (int i = t; i < Nn * Nn; i += 256) adjs[i] = adj[i];
  __syncthreads();

  const int b = blockIdx.y;
  const int k = blockIdx.x * 256 + t;
  const size_t base = (size_t)b * Nn * D + k;

  float v[Nn];
  #pragma unroll
  for (int j = 0; j < Nn; ++j) v[j] = bf2f(h1[base + (size_t)j * D]);
  #pragma unroll
  for (int i = 0; i < Nn; ++i) {
    float a = 0.f;
    #pragma unroll
    for (int j = 0; j < Nn; ++j) a += adjs[i * Nn + j] * v[j];
    a = (a > 0.f) ? a : 0.2f * a;
    xout[base + (size_t)i * D] = f2bf(a);
  }
}

// ---------------------------------------------------------------------------
// Fused tail, one block per batch b:
//   x2 = adj @ h2 ; scores = F x2^T ; softmax ; anatomy = attn@F + F ;
//   LayerNorm ; node = anatomy@fcn_w^T + fcn_b ; graph = mean@fcg_w^T + fcg_b
// ---------------------------------------------------------------------------
__global__ __launch_bounds__(256)
void tail_kernel(const float* __restrict__ feature,
                 const unsigned short* __restrict__ h2,
                 const float* __restrict__ adj,
                 const float* __restrict__ ln_w, const float* __restrict__ ln_b,
                 const float* __restrict__ fcn_w, const float* __restrict__ fcn_b,
                 const float* __restrict__ fcg_w, const float* __restrict__ fcg_b,
                 float* __restrict__ node_out, float* __restrict__ graph_out)
{
  constexpr int D = D_OUT;
  __shared__ float Fs[Nn * D];   // 72 KB  feature[b]
  __shared__ float A2[Nn * D];   // 72 KB  x2, then anatomy (reused)
  __shared__ float adjs[Nn * Nn];
  __shared__ float Sc[Nn * Nn];  // scores -> attn
  __shared__ float g[D];         // graph mean

  const int t = threadIdx.x;
  const int b = blockIdx.x;
  const int wave = t >> 5, lane = t & 31;

  for (int i = t; i < Nn * Nn; i += 256) adjs[i] = adj[i];
  const float* F = feature + (size_t)b * Nn * D;
  for (int i = t; i < Nn * D; i += 256) Fs[i] = F[i];
  __syncthreads();

  // x2 = adj @ h2[b]
  const unsigned short* H = h2 + (size_t)b * Nn * D;
  for (int k = t; k < D; k += 256) {
    float v[Nn];
    #pragma unroll
    for (int j = 0; j < Nn; ++j) v[j] = bf2f(H[j * D + k]);
    #pragma unroll
    for (int i = 0; i < Nn; ++i) {
      float a = 0.f;
      #pragma unroll
      for (int j = 0; j < Nn; ++j) a += adjs[i * Nn + j] * v[j];
      A2[i * D + k] = a;
    }
  }
  __syncthreads();

  // scores[n,m] = F[n] . x2[m]   (wave per pair, lane-strided dot)
  for (int p = wave; p < Nn * Nn; p += 8) {
    const int n = p / Nn, m = p % Nn;
    float s = 0.f;
    for (int d = lane; d < D; d += 32) s += Fs[n * D + d] * A2[m * D + d];
    s = waveReduce(s);
    if (lane == 0) Sc[p] = s;
  }
  __syncthreads();

  // softmax over m (one thread per row; 18x18, trivial)
  if (t < Nn) {
    float mx = -3.4e38f;
    for (int m = 0; m < Nn; ++m) mx = fmaxf(mx, Sc[t * Nn + m]);
    float sm = 0.f;
    for (int m = 0; m < Nn; ++m) { float e = expf(Sc[t * Nn + m] - mx); Sc[t * Nn + m] = e; sm += e; }
    float inv = 1.f / sm;
    for (int m = 0; m < Nn; ++m) Sc[t * Nn + m] *= inv;
  }
  __syncthreads();

  // anatomy = attn @ F + F   (overwrite A2; x2 fully consumed)
  for (int idx = t; idx < Nn * D; idx += 256) {
    const int n = idx >> 10, d = idx & (D - 1);
    float a = Fs[idx];
    #pragma unroll
    for (int m = 0; m < Nn; ++m) a += Sc[n * Nn + m] * Fs[m * D + d];
    A2[idx] = a;
  }
  __syncthreads();

  // LayerNorm per row (wave per row)
  for (int n = wave; n < Nn; n += 8) {
    float s = 0.f, s2 = 0.f;
    for (int d = lane; d < D; d += 32) { float x = A2[n * D + d]; s += x; s2 += x * x; }
    s = waveReduce(s); s2 = waveReduce(s2);
    const float mu = s * (1.f / D);
    const float var = s2 * (1.f / D) - mu * mu;
    const float rs = rsqrtf(var + 1e-5f);
    for (int d = lane; d < D; d += 32)
      A2[n * D + d] = (A2[n * D + d] - mu) * rs * ln_w[d] + ln_b[d];
  }
  __syncthreads();

  // node_level[b,n,c]
  for (int p = wave; p < Nn * NCLS; p += 8) {
    const int n = p / NCLS, c = p % NCLS;
    float s = 0.f;
    for (int d = lane; d < D; d += 32) s += A2[n * D + d] * fcn_w[c * D + d];
    s = waveReduce(s);
    if (lane == 0) node_out[((size_t)b * Nn + n) * NCLS + c] = s + fcn_b[c];
  }

  // graph mean then fcg
  for (int d = t; d < D; d += 256) {
    float s = 0.f;
    #pragma unroll
    for (int n = 0; n < Nn; ++n) s += A2[n * D + d];
    g[d] = s * (1.f / Nn);
  }
  __syncthreads();
  for (int c = wave; c < NCLS; c += 8) {
    float s = 0.f;
    for (int d = lane; d < D; d += 32) s += g[d] * fcg_w[c * D + d];
    s = waveReduce(s);
    if (lane == 0) graph_out[(size_t)b * NCLS + c] = s + fcg_b[c];
  }
}

// ---------------------------------------------------------------------------
// Launch: convert feature/w1/w2 -> GEMM1 -> adj+LeakyReLU -> GEMM2 -> tail
// Workspace: featB 75.5MB | w1t 4MB | w2t 4MB | bufA 151MB | bufB 151MB
// ---------------------------------------------------------------------------
extern "C" void kernel_launch(void* const* d_in, const int* in_sizes, int n_in,
                              void* d_out, int out_size, void* d_ws, size_t ws_size,
                              hipStream_t stream)
{
  const float* feature = (const float*)d_in[0];
  const float* adj     = (const float*)d_in[1];
  const float* w1      = (const float*)d_in[2];
  const float* w2      = (const float*)d_in[3];
  const float* ln_w    = (const float*)d_in[4];
  const float* ln_b    = (const float*)d_in[5];
  const float* fcn_w   = (const float*)d_in[6];
  const float* fcn_b   = (const float*)d_in[7];
  const float* fcg_w   = (const float*)d_in[8];
  const float* fcg_b   = (const float*)d_in[9];

  float* node_out  = (float*)d_out;                      // [B,N,C]
  float* graph_out = node_out + (size_t)Bsz * Nn * NCLS; // [B,C]

  unsigned short* featB = (unsigned short*)d_ws;                 // [M, 1024] bf16
  unsigned short* w1t   = featB + (size_t)Mrows * D_IN;          // [2048,1024] bf16
  unsigned short* w2t   = w1t   + (size_t)D_HID * D_IN;          // [1024,2048] bf16
  unsigned short* bufA  = w2t   + (size_t)D_OUT * D_HID;         // h1 then h2 bf16
  unsigned short* bufB  = bufA  + (size_t)Mrows * D_HID;         // x bf16

  // 0) precision conversions (one pass each; removes all cvt work from GEMMs)
  {
    const long long n8 = (long long)Mrows * D_IN / 8;
    convert_bf16_kernel<<<(unsigned)((n8 + 255) / 256), 256, 0, stream>>>(feature, featB, n8);
    dim3 g1(D_HID / 32, D_IN / 32);   // w1 [1024,2048] -> w1t [2048,1024]
    transpose_convert_kernel<<<g1, 256, 0, stream>>>(w1, w1t, D_IN, D_HID);
    dim3 g2(D_OUT / 32, D_HID / 32);  // w2 [2048,1024] -> w2t [1024,2048]
    transpose_convert_kernel<<<g2, 256, 0, stream>>>(w2, w2t, D_HID, D_OUT);
  }
  // 1) h1 = feature @ w1   [36864,1024] x [1024,2048] -> bf16
  {
    dim3 grid(D_HID / 128, Mrows / 128);  // (16, 288)
    gemm_wmma_bf16<<<grid, 256, 0, stream>>>(featB, w1t, bufA, Mrows, D_HID, D_IN);
  }
  // 2) x = LeakyReLU(adj @ h1)  -> bf16
  {
    dim3 grid(D_HID / 256, Bsz);
    adj_lrelu_kernel<<<grid, 256, 0, stream>>>(bufA, adj, bufB, D_HID);
  }
  // 3) h2 = x @ w2   [36864,2048] x [2048,1024] -> bf16 (reuse bufA)
  {
    dim3 grid(D_OUT / 128, Mrows / 128);  // (8, 288)
    gemm_wmma_bf16<<<grid, 256, 0, stream>>>(bufB, w2t, bufA, Mrows, D_OUT, D_HID);
  }
  // 4) fused attention/LN/heads tail, one block per batch
  tail_kernel<<<Bsz, 256, 0, stream>>>(feature, bufA, adj, ln_w, ln_b,
                                       fcn_w, fcn_b, fcg_w, fcg_b,
                                       node_out, graph_out);
}
